// Mamba3Block_71356586655993
// MI455X (gfx1250) — matrix operand order repaired and checked by reading, not compiled, not run on hardware
//
#include <hip/hip_runtime.h>
#include <hip/hip_bf16.h>

// ---------------------------------------------------------------------------
// Mamba block for MI455X (gfx1250), bf16 WMMA GEMMs + chunked selective scan.
// ---------------------------------------------------------------------------

typedef __bf16 bf16_t;
typedef __attribute__((ext_vector_type(16))) bf16_t v16bf;
typedef __attribute__((ext_vector_type(8)))  float  v8f;

union FragAB {
    uint4 q[2];
    v16bf v;
};
static_assert(sizeof(FragAB) == 32, "frag size");

#define MB_B 4
#define MB_L 4096
#define MB_D 1024
#define MB_M (MB_B * MB_L)   // 16384
#define MB_K 1024
#define NCHUNK 16
#define CLEN   256           // NCHUNK * CLEN == MB_L

enum { MODE_PLAIN = 0, MODE_DBC = 1, MODE_OUT = 2 };

__device__ __forceinline__ unsigned short f2bf(float f) {
    unsigned u = __float_as_uint(f);
    unsigned r = u + 0x7FFFu + ((u >> 16) & 1u);   // round-to-nearest-even
    return (unsigned short)(r >> 16);
}

__device__ __forceinline__ float wave_sum(float v) {
    #pragma unroll
    for (int o = 16; o > 0; o >>= 1) v += __shfl_xor(v, o, 32);
    return v;
}

// --------------------------- fp32 -> bf16 copy ------------------------------
__global__ void convert_f32_bf16(const float* __restrict__ src,
                                 unsigned short* __restrict__ dst, int n) {
    int i = blockIdx.x * 256 + threadIdx.x;
    if (i < n) dst[i] = f2bf(src[i]);
}

// ------------------------------ LayerNorm -----------------------------------
__global__ __launch_bounds__(256) void layernorm_kernel(
    const float* __restrict__ x, const float* __restrict__ g,
    const float* __restrict__ bta, unsigned short* __restrict__ hbf) {
    const int D = MB_D;
    int m = blockIdx.x;
    const float4* row = (const float4*)(x + (size_t)m * D);
    float4 v = row[threadIdx.x];                      // 256 threads * 4 = 1024
    float s  = v.x + v.y + v.z + v.w;
    float ss = v.x*v.x + v.y*v.y + v.z*v.z + v.w*v.w;
    s  = wave_sum(s);
    ss = wave_sum(ss);
    __shared__ float sh[16];
    int wave = threadIdx.x >> 5, lane = threadIdx.x & 31;
    if (lane == 0) { sh[wave] = s; sh[8 + wave] = ss; }
    __syncthreads();
    if (wave == 0) {
        float a  = (lane < 8) ? sh[lane]     : 0.f;
        float b2 = (lane < 8) ? sh[8 + lane] : 0.f;
        a  = wave_sum(a);
        b2 = wave_sum(b2);
        if (lane == 0) { sh[0] = a; sh[1] = b2; }
    }
    __syncthreads();
    float mu   = sh[0] * (1.f / D);
    float var  = sh[1] * (1.f / D) - mu * mu;
    float rstd = rsqrtf(var + 1e-5f);
    int d0 = threadIdx.x * 4;
    ushort4 o;
    o.x = f2bf((v.x - mu) * rstd * g[d0 + 0] + bta[d0 + 0]);
    o.y = f2bf((v.y - mu) * rstd * g[d0 + 1] + bta[d0 + 1]);
    o.z = f2bf((v.z - mu) * rstd * g[d0 + 2] + bta[d0 + 2]);
    o.w = f2bf((v.w - mu) * rstd * g[d0 + 3] + bta[d0 + 3]);
    ((ushort4*)(hbf + (size_t)m * D))[threadIdx.x] = o;
}

// ------------------------------ WMMA GEMM -----------------------------------
// C[M,N] = A[M,K](bf16) * Bw[N,K]^T(bf16), f32 accumulate, fused epilogues.
// Block: 256 threads = 8 waves, 128x128 tile; wave tile 64x32 (4x2 wmma).
__global__ __launch_bounds__(256) void gemm_bf16_wmma(
    const bf16_t* __restrict__ A, const bf16_t* __restrict__ Bw,
    int M, int N, int K, int mode,
    const float* __restrict__ bias0, const float* __restrict__ bias1,
    const float* __restrict__ bias2, const float* __restrict__ aux,
    float* __restrict__ out0, float* __restrict__ out1,
    float* __restrict__ out2) {
    const int lane = threadIdx.x & 31;
    const int wave = threadIdx.x >> 5;
    const int wm = wave >> 2;                 // 0..1
    const int wn = wave & 3;                  // 0..3
    const int m0 = blockIdx.y * 128 + wm * 64;
    const int n0 = blockIdx.x * 128 + wn * 32;

    const v8f vzero = {0.f,0.f,0.f,0.f,0.f,0.f,0.f,0.f};
    v8f acc[4][2];
    #pragma unroll
    for (int i = 0; i < 4; ++i)
        #pragma unroll
        for (int j = 0; j < 2; ++j) acc[i][j] = vzero;

    const int ar = lane & 15;
    const int ak = (lane >> 4) * 16;   // A: lanes 0-15 K 0..15; 16-31 K 16..31
    const int bk = (lane >> 4) * 16;   // B: lanes 0-15 K 0..15; 16-31 K 16..31

    for (int k0 = 0; k0 < K; k0 += 32) {
        FragAB a[4], b[2];
        #pragma unroll
        for (int i = 0; i < 4; ++i) {
            const bf16_t* p = A + (size_t)(m0 + i * 16 + ar) * K + (k0 + ak);
            a[i].q[0] = *(const uint4*)(p);
            a[i].q[1] = *(const uint4*)(p + 8);
        }
        #pragma unroll
        for (int j = 0; j < 2; ++j) {
            const bf16_t* p = Bw + (size_t)(n0 + j * 16 + ar) * K + (k0 + bk);
            b[j].q[0] = *(const uint4*)(p);
            b[j].q[1] = *(const uint4*)(p + 8);
        }
        #pragma unroll
        for (int i = 0; i < 4; ++i)
            #pragma unroll
            for (int j = 0; j < 2; ++j)
                acc[i][j] = __builtin_amdgcn_wmma_f32_16x16x32_bf16(
                    false, a[i].v, false, b[j].v, (short)0, acc[i][j],
                    false, false);
    }

    // C/D layout: lane&15 = column, (lane>>4)*8 + vgpr = row.
    const int colb = lane & 15;
    const int rowb = (lane >> 4) * 8;
    const int seg  = n0 >> 10;   // which 1024-wide segment (DBC only)
    #pragma unroll
    for (int i = 0; i < 4; ++i) {
        #pragma unroll
        for (int j = 0; j < 2; ++j) {
            int n = n0 + j * 16 + colb;
            #pragma unroll
            for (int v = 0; v < 8; ++v) {
                int m = m0 + i * 16 + rowb + v;
                float val = acc[i][j][v];
                if (mode == MODE_PLAIN) {
                    out0[(size_t)m * N + n] = val + bias0[n];
                } else if (mode == MODE_OUT) {
                    size_t idx = (size_t)m * N + n;
                    out0[idx] = val + bias0[n] + aux[idx];     // + residual x
                } else {  // MODE_DBC
                    if (seg == 0) {
                        float xv = val + bias0[n];
                        // exp(-softplus(xv)) == sigmoid(-xv)
                        out0[(size_t)m * 1024 + n] = 1.0f / (1.0f + __expf(xv));
                    } else if (seg == 1) {
                        int nn = n - 1024;
                        float bm = val + bias1[nn];
                        out1[(size_t)m * 1024 + nn] =
                            aux[(size_t)m * 1024 + nn] * bm;    // u_c * B
                    } else {
                        int nn = n - 2048;
                        out2[(size_t)m * 1024 + nn] = val + bias2[nn];
                    }
                }
            }
        }
    }
}

// --------------------- depthwise causal conv (K=4) + bf16 -------------------
__global__ void conv_kernel(const float* __restrict__ uv,
                            const float* __restrict__ wconv,
                            const float* __restrict__ bconv,
                            float* __restrict__ uc,
                            unsigned short* __restrict__ ucbf) {
    size_t idx = (size_t)blockIdx.x * 256 + threadIdx.x;  // over M*D
    int d = (int)(idx % MB_D);
    size_t m = idx / MB_D;
    int t  = (int)(m % MB_L);
    int bb = (int)(m / MB_L);
    float acc = bconv[d];
    #pragma unroll
    for (int k = 0; k < 4; ++k) {
        int tt = t - 3 + k;
        if (tt >= 0)
            acc += wconv[d * 4 + k] *
                   uv[((size_t)bb * MB_L + tt) * 2048 + d];  // u = cols [0,1024)
    }
    uc[idx]   = acc;
    ucbf[idx] = f2bf(acc);
}

// ------------------------- chunked selective scan ---------------------------
// Wave owns (b, 128-wide n-slice, chunk); lane owns 4 consecutive n.
__global__ __launch_bounds__(256) void scan_phase1(
    const float* __restrict__ decay, const float* __restrict__ inp,
    float* __restrict__ S, float* __restrict__ P) {
    const int NB = MB_D / 128;  // 8
    int wave = threadIdx.x >> 5, lane = threadIdx.x & 31;
    int gw = blockIdx.x * 8 + wave;                 // 0..511
    int ch = gw % NCHUNK;
    int r  = gw / NCHUNK;
    int nb = r % NB;
    int bb = r / NB;
    int n0 = nb * 128 + lane * 4;
    size_t base = ((size_t)bb * MB_L + (size_t)ch * CLEN) * MB_D + n0;
    float4 h = {0.f, 0.f, 0.f, 0.f};
    float4 p = {1.f, 1.f, 1.f, 1.f};
    for (int t = 0; t < CLEN; ++t) {
        float4 dd = *(const float4*)(decay + base + (size_t)t * MB_D);
        float4 ii = *(const float4*)(inp   + base + (size_t)t * MB_D);
        h.x = h.x * dd.x + ii.x;  p.x *= dd.x;
        h.y = h.y * dd.y + ii.y;  p.y *= dd.y;
        h.z = h.z * dd.z + ii.z;  p.z *= dd.z;
        h.w = h.w * dd.w + ii.w;  p.w *= dd.w;
    }
    size_t sidx = ((size_t)(bb * NCHUNK + ch)) * MB_D + n0;
    *(float4*)(S + sidx) = h;
    *(float4*)(P + sidx) = p;
}

__global__ void scan_combine(const float* __restrict__ S,
                             const float* __restrict__ P,
                             float* __restrict__ Hst) {
    int i = blockIdx.x * 256 + threadIdx.x;          // over B*N
    int bb = i / MB_D, nn = i % MB_D;
    float h = 0.f;
    for (int c = 0; c < NCHUNK; ++c) {
        size_t id = ((size_t)(bb * NCHUNK + c)) * MB_D + nn;
        Hst[id] = h;
        h = h * P[id] + S[id];
    }
}

__global__ __launch_bounds__(256) void scan_phase2(
    const float* __restrict__ decay, const float* __restrict__ inp,
    const float* __restrict__ Cm, const float* __restrict__ Hst,
    float* __restrict__ y) {
    const int NB = MB_D / 128;
    int wave = threadIdx.x >> 5, lane = threadIdx.x & 31;
    int gw = blockIdx.x * 8 + wave;
    int ch = gw % NCHUNK;
    int r  = gw / NCHUNK;
    int nb = r % NB;
    int bb = r / NB;
    int n0 = nb * 128 + lane * 4;
    size_t base = ((size_t)bb * MB_L + (size_t)ch * CLEN) * MB_D + n0;
    size_t sidx = ((size_t)(bb * NCHUNK + ch)) * MB_D + n0;
    float4 h = *(const float4*)(Hst + sidx);
    float* yb = y + (size_t)bb * MB_L + (size_t)ch * CLEN;
    for (int t = 0; t < CLEN; ++t) {
        float4 dd = *(const float4*)(decay + base + (size_t)t * MB_D);
        float4 ii = *(const float4*)(inp   + base + (size_t)t * MB_D);
        float4 cc = *(const float4*)(Cm    + base + (size_t)t * MB_D);
        h.x = h.x * dd.x + ii.x;
        h.y = h.y * dd.y + ii.y;
        h.z = h.z * dd.z + ii.z;
        h.w = h.w * dd.w + ii.w;
        float val = h.x * cc.x + h.y * cc.y + h.z * cc.z + h.w * cc.w;
        val = wave_sum(val);
        if (lane == 0) atomicAdd(yb + t, val);
    }
}

// --------------------------- gate: y * sigmoid(v) ---------------------------
__global__ void gate_kernel(const float* __restrict__ uv,
                            const float* __restrict__ y,
                            unsigned short* __restrict__ ybf) {
    size_t idx = (size_t)blockIdx.x * 256 + threadIdx.x;  // over M*D
    int d = (int)(idx % MB_D);
    size_t m = idx / MB_D;
    float v  = uv[m * 2048 + 1024 + d];                   // v = cols [1024,2048)
    float sg = 1.0f / (1.0f + __expf(-v));
    ybf[idx] = f2bf(y[m] * sg);
}

// ------------------------------- launcher -----------------------------------
extern "C" void kernel_launch(void* const* d_in, const int* in_sizes, int n_in,
                              void* d_out, int out_size, void* d_ws,
                              size_t ws_size, hipStream_t stream) {
    const float* x       = (const float*)d_in[0];
    const float* ln_g    = (const float*)d_in[1];
    const float* ln_b    = (const float*)d_in[2];
    const float* w_in    = (const float*)d_in[3];
    const float* b_in    = (const float*)d_in[4];
    const float* w_conv  = (const float*)d_in[5];
    const float* b_conv  = (const float*)d_in[6];
    const float* w_delta = (const float*)d_in[7];
    const float* b_delta = (const float*)d_in[8];
    const float* w_B     = (const float*)d_in[9];
    const float* b_B     = (const float*)d_in[10];
    const float* w_C     = (const float*)d_in[11];
    const float* b_C     = (const float*)d_in[12];
    const float* w_out   = (const float*)d_in[13];
    const float* b_out   = (const float*)d_in[14];
    float* out = (float*)d_out;

    const int M = MB_M, D = MB_D, K = MB_K;

    // Workspace layout (256B aligned slices)
    char* ws = (char*)d_ws;
    size_t off = 0;
    auto take = [&](size_t bytes) {
        char* p = ws + off;
        off += (bytes + 255) & ~(size_t)255;
        return p;
    };
    unsigned short* h_bf    = (unsigned short*)take((size_t)M * D * 2);
    unsigned short* win_bf  = (unsigned short*)take((size_t)2048 * K * 2);
    unsigned short* wdbc_bf = (unsigned short*)take((size_t)3072 * K * 2);
    unsigned short* wout_bf = (unsigned short*)take((size_t)1024 * K * 2);
    float*          uv      = (float*)take((size_t)M * 2048 * 4);
    float*          uc      = (float*)take((size_t)M * D * 4);
    unsigned short* uc_bf   = (unsigned short*)take((size_t)M * D * 2);
    float*          decay   = (float*)take((size_t)M * D * 4);
    float*          inp     = (float*)take((size_t)M * D * 4);
    float*          Cmat    = (float*)take((size_t)M * D * 4);
    unsigned short* ybar_bf = (unsigned short*)take((size_t)M * D * 2);
    float*          yv      = (float*)take((size_t)M * 4);
    float*          Ssc     = (float*)take((size_t)MB_B * NCHUNK * D * 4);
    float*          Psc     = (float*)take((size_t)MB_B * NCHUNK * D * 4);
    float*          Hst     = (float*)take((size_t)MB_B * NCHUNK * D * 4);
    if (off > ws_size) return;  // workspace too small: bail deterministically

    // 1) weights -> bf16 (w_delta|w_B|w_C concatenated rows -> [3072,1024])
    {
        int n1 = 2048 * K, n2 = 1024 * K;
        convert_f32_bf16<<<(n1 + 255) / 256, 256, 0, stream>>>(w_in, win_bf, n1);
        convert_f32_bf16<<<(n2 + 255) / 256, 256, 0, stream>>>(w_delta, wdbc_bf, n2);
        convert_f32_bf16<<<(n2 + 255) / 256, 256, 0, stream>>>(w_B, wdbc_bf + (size_t)1024 * K, n2);
        convert_f32_bf16<<<(n2 + 255) / 256, 256, 0, stream>>>(w_C, wdbc_bf + (size_t)2048 * K, n2);
        convert_f32_bf16<<<(n2 + 255) / 256, 256, 0, stream>>>(w_out, wout_bf, n2);
    }

    // 2) LayerNorm -> h (bf16)
    layernorm_kernel<<<M, 256, 0, stream>>>(x, ln_g, ln_b, h_bf);

    // 3) uv = h @ w_in^T + b_in
    gemm_bf16_wmma<<<dim3(2048 / 128, M / 128), 256, 0, stream>>>(
        (const bf16_t*)h_bf, (const bf16_t*)win_bf, M, 2048, K, MODE_PLAIN,
        b_in, nullptr, nullptr, nullptr, uv, nullptr, nullptr);

    // 4) depthwise causal conv + bf16 emit
    conv_kernel<<<(int)(((size_t)M * D) / 256), 256, 0, stream>>>(
        uv, w_conv, b_conv, uc, uc_bf);

    // 5) [delta|B|C] = u_c @ [w_delta;w_B;w_C]^T ; fused epilogue -> decay, inp, C
    gemm_bf16_wmma<<<dim3(3072 / 128, M / 128), 256, 0, stream>>>(
        (const bf16_t*)uc_bf, (const bf16_t*)wdbc_bf, M, 3072, K, MODE_DBC,
        b_delta, b_B, b_C, uc, decay, inp, Cmat);

    // 6) selective scan (chunked 3-phase)
    hipMemsetAsync(yv, 0, (size_t)M * 4, stream);
    {
        int nwaves = MB_B * (D / 128) * NCHUNK;   // 512
        scan_phase1<<<nwaves / 8, 256, 0, stream>>>(decay, inp, Ssc, Psc);
        scan_combine<<<(MB_B * D) / 256, 256, 0, stream>>>(Ssc, Psc, Hst);
        scan_phase2<<<nwaves / 8, 256, 0, stream>>>(decay, inp, Cmat, Hst, yv);
    }

    // 7) ybar = y * sigmoid(v) (bf16)
    gate_kernel<<<(int)(((size_t)M * D) / 256), 256, 0, stream>>>(uv, yv, ybar_bf);

    // 8) out = x + ybar @ w_out^T + b_out
    gemm_bf16_wmma<<<dim3(1024 / 128, M / 128), 256, 0, stream>>>(
        (const bf16_t*)ybar_bf, (const bf16_t*)wout_bf, M, 1024, K, MODE_OUT,
        b_out, nullptr, nullptr, x, out, nullptr, nullptr);
}
